// Retinanet2_55594056679958
// MI455X (gfx1250) — matrix operand order repaired and checked by reading, not compiled, not run on hardware
//
#include <hip/hip_runtime.h>
#include <stdint.h>

// ---------------- CDNA5 WMMA types ----------------
typedef __attribute__((ext_vector_type(16))) __bf16 v16bf;
typedef __attribute__((ext_vector_type(8)))  float  v8f;

#define NB      4
#define NCLS    80
#define NA      9
#define KTOT    2304      // 9 taps * 256 input channels
#define TOPK_N  1000
#define NLEV    5
#define NCAND   5000
#define SORT_N  8192
#define MAXPHW  (82 * 82) // max padded H*W (p3)

union FragBF { uint4 q[2]; v16bf v; };

__device__ __forceinline__ unsigned short f2bf(float f) {
  unsigned u = __float_as_uint(f);
  u += 0x7FFFu + ((u >> 16) & 1u);          // round-to-nearest-even
  return (unsigned short)(u >> 16);
}

// ---------------- layout conversion ----------------
// NCHW fp32 -> padded NHWC bf16 [B][H+2][W+2][256], halo = 0
__global__ void nchw_to_padded(const float* __restrict__ in,
                               unsigned short* __restrict__ out, int H, int W) {
  int PW = W + 2, PH = H + 2;
  size_t i = (size_t)blockIdx.x * blockDim.x + threadIdx.x;
  size_t total = (size_t)NB * PH * PW * 256;
  if (i >= total) return;
  int c = (int)(i & 255);
  size_t t = i >> 8;
  int pp = (int)(t % (size_t)(PH * PW));
  int b  = (int)(t / (size_t)(PH * PW));
  int py = pp / PW, px = pp - py * PW;
  float v = 0.f;
  if (py >= 1 && py <= H && px >= 1 && px <= W) {
    int m = (py - 1) * W + (px - 1);
    v = in[((size_t)b * 256 + c) * (size_t)(H * W) + m];
  }
  out[i] = f2bf(v);
}

// [Cout][Cin][3][3] fp32 -> bf16 rows [Cout][tap*256+ci]
__global__ void pack_weights(const float* __restrict__ w,
                             unsigned short* __restrict__ dst, int cout) {
  size_t i = (size_t)blockIdx.x * blockDim.x + threadIdx.x;
  size_t total = (size_t)cout * KTOT;
  if (i >= total) return;
  int k  = (int)(i % KTOT);
  int co = (int)(i / KTOT);
  int tap = k >> 8, ci = k & 255;
  dst[i] = f2bf(w[((size_t)co * 256 + ci) * 9 + tap]);
}

__global__ void zero_u32(unsigned* p, size_t n) {
  for (size_t i = (size_t)blockIdx.x * blockDim.x + threadIdx.x; i < n;
       i += (size_t)gridDim.x * blockDim.x)
    p[i] = 0u;
}

// ---------------- implicit-GEMM conv3x3 via WMMA ----------------
// Input: padded NHWC bf16. M = H*W (per image), N = Cout, K = 2304.
// One wave -> one 16x16 tile; inner loop: 9 taps x 8 unconditional K-steps.
// mode 0: bias+ReLU -> padded bf16 NHWC (Cout==256)
// mode 1: bias+sigmoid -> f32 [b][HW][Cout]   mode 2: bias -> f32
__global__ __launch_bounds__(256) void conv3x3_wmma(
    const unsigned short* __restrict__ act, const unsigned short* __restrict__ wt,
    const float* __restrict__ bias, unsigned short* __restrict__ outBf,
    float* __restrict__ outF, int H, int W, int Cout, int nTiles, int mode) {
  int wave  = threadIdx.x >> 5;
  int nTile = blockIdx.y * 8 + wave;
  if (nTile >= nTiles) return;                 // wave-uniform

  int lane = threadIdx.x & 31;
  int half = lane >> 4;                        // K-half select (WMMA layout)
  int row  = lane & 15;
  int b  = blockIdx.z;
  int HW = H * W, PW = W + 2, PH = H + 2;
  int m  = blockIdx.x * 16 + row;              // spatial row of A
  int mc = (m < HW) ? m : (HW - 1);            // clamp: loads stay in-bounds
  int y = mc / W, x = mc - y * W;
  int n  = nTile * 16 + row;                   // output channel of B
  int nc = (n < Cout) ? n : 0;

  const uint4* actq = (const uint4*)act;       // 8 bf16 per uint4
  const uint4* wq   = (const uint4*)(wt + (size_t)nc * KTOT);
  size_t center = (size_t)b * PH * PW + (size_t)(y + 1) * PW + (x + 1);

  v8f acc = {};
  for (int tap = 0; tap < 9; ++tap) {
    int ty = tap / 3, tx = tap - ty * 3;
    const uint4* ap = actq + ((center + (size_t)(ty - 1) * PW + (tx - 1)) << 5);
    const uint4* wp = wq + tap * 32;           // tap*256 channels / 8
    if (tap < 8) __builtin_prefetch(wp + 32, 0, 0);
#pragma unroll
    for (int kc = 0; kc < 8; ++kc) {           // 32 channels per step
      FragBF a, bb;
      a.q[0]  = ap[kc * 4 + half];             // ci = kc*32 + half*8
      a.q[1]  = ap[kc * 4 + half + 2];         // ci + 16
      bb.q[0] = wp[kc * 4 + half];             // kg = tap*256 + kc*32 + half*8
      bb.q[1] = wp[kc * 4 + half + 2];         // kg + 16
      acc = __builtin_amdgcn_wmma_f32_16x16x32_bf16(
          false, a.v, false, bb.v, (short)0, acc, false, false);
    }
  }

  bool vn = (n < Cout);
  float bv = vn ? bias[n] : 0.f;
#pragma unroll
  for (int r = 0; r < 8; ++r) {
    int mg = blockIdx.x * 16 + r + half * 8;   // C layout: lanes>=16 hold M=r+8
    if (mg < HW && vn) {
      float v = acc[r] + bv;
      if (mode == 0) {
        v = v > 0.f ? v : 0.f;
        int gy = mg / W, gx = mg - gy * W;
        outBf[((size_t)b * PH * PW + (size_t)(gy + 1) * PW + (gx + 1)) * 256 + n] =
            f2bf(v);
      } else if (mode == 1) {
        outF[((size_t)b * HW + mg) * (size_t)Cout + n] = 1.f / (1.f + __expf(-v));
      } else {
        outF[((size_t)b * HW + mg) * (size_t)Cout + n] = v;
      }
    }
  }
}

// ---------------- top-k selection (radix-ish on float bits) ----------------
__global__ __launch_bounds__(1024) void hist_topk(const float* __restrict__ prob,
                                                  int N, int* __restrict__ tg) {
  __shared__ int hist[1024];
  int b = blockIdx.x;
  for (int i = threadIdx.x; i < 1024; i += blockDim.x) hist[i] = 0;
  __syncthreads();
  const float* p = prob + (size_t)b * N;
  for (int i = threadIdx.x; i < N; i += blockDim.x) {
    unsigned bin = __float_as_uint(p[i]) >> 21;   // scores in (0,1): bin < 512
    atomicAdd(&hist[bin < 1024u ? bin : 1023u], 1);
  }
  __syncthreads();
  if (threadIdx.x == 0) {
    int acc = 0, T = 0, G = 0;
    for (int bin = 1023; bin >= 0; --bin) {
      int c = hist[bin];
      if (acc + c >= TOPK_N) { T = bin; G = acc; break; }
      acc += c;
    }
    tg[b * 2] = T; tg[b * 2 + 1] = G;
  }
}

__global__ void topk_compact(const float* __restrict__ prob, const int* __restrict__ tg,
                             int* __restrict__ cnt, float* __restrict__ candS,
                             int* __restrict__ candI, int N, int lev) {
  int b = blockIdx.y;
  int T = tg[b * 2], G = tg[b * 2 + 1];
  const float* p = prob + (size_t)b * N;
  for (int i = blockIdx.x * blockDim.x + threadIdx.x; i < N;
       i += gridDim.x * blockDim.x) {
    float s = p[i];
    int bin = (int)(__float_as_uint(s) >> 21);
    int pos = -1;
    if (bin > T) pos = atomicAdd(&cnt[b * 2], 1);
    else if (bin == T) {
      int q = G + atomicAdd(&cnt[b * 2 + 1], 1);
      if (q < TOPK_N) pos = q;
    }
    if (pos >= 0 && pos < TOPK_N) {
      size_t slot = (size_t)b * NCAND + lev * TOPK_N + pos;
      candS[slot] = s;
      candI[slot] = i;
    }
  }
}

// ---------------- anchors + decode ----------------
__global__ void decode_boxes(const int* __restrict__ candI, const float* __restrict__ reg,
                             float* __restrict__ candBox, int* __restrict__ candC,
                             int H, int W, int strideLv, int lev) {
  int t = blockIdx.x * blockDim.x + threadIdx.x;
  if (t >= NB * TOPK_N) return;
  int b = t / TOPK_N, j = t - b * TOPK_N;
  size_t slot = (size_t)b * NCAND + lev * TOPK_N + j;
  int idx = candI[slot];
  int c  = idx % NCLS;
  int ai = idx / NCLS;               // anchor index = m*9 + a
  int m = ai / NA, a = ai - m * NA;
  int yy = m / W, xx = m - yy * W;
  int ri = a / 3, si = a - ri * 3;
  float s = (float)strideLv, size = 4.f * s;
  float scale = exp2f((float)si * (1.f / 3.f));
  float ratio = exp2f((float)(ri - 1));          // 0.5, 1, 2
  float aw = size * scale * sqrtf(1.f / ratio);
  float ah = size * scale * sqrtf(ratio);
  float cx = (xx + 0.5f) * s, cy = (yy + 0.5f) * s;
  const float* d = reg + ((size_t)b * H * W + m) * 36 + a * 4;
  float ncx = cx + d[0] * aw, ncy = cy + d[1] * ah;
  float nw = aw * __expf(d[2]), nh = ah * __expf(d[3]);
  candBox[slot * 4 + 0] = ncx - 0.5f * nw;
  candBox[slot * 4 + 1] = ncy - 0.5f * nh;
  candBox[slot * 4 + 2] = ncx + 0.5f * nw;
  candBox[slot * 4 + 3] = ncy + 0.5f * nh;
  candC[slot] = c;
}

// ---------------- bitonic sort (desc) of 5000 padded to 8192, one block/image ----
__global__ __launch_bounds__(1024) void sort_desc(const float* __restrict__ candS,
                                                  float* __restrict__ sScore,
                                                  int* __restrict__ sIdx) {
  extern __shared__ unsigned char smraw[];
  float* key = (float*)smraw;
  int*   val = (int*)(smraw + SORT_N * sizeof(float));
  int b = blockIdx.x;
  for (int i = threadIdx.x; i < SORT_N; i += blockDim.x) {
    key[i] = (i < NCAND) ? candS[(size_t)b * NCAND + i] : -1e30f;
    val[i] = i;
  }
  __syncthreads();
  for (int k = 2; k <= SORT_N; k <<= 1) {
    for (int j = k >> 1; j > 0; j >>= 1) {
      for (int i = threadIdx.x; i < SORT_N; i += blockDim.x) {
        int ixj = i ^ j;
        if (ixj > i) {
          bool desc = ((i & k) == 0);
          float ki = key[i], kj = key[ixj];
          bool sw = desc ? (ki < kj) : (ki > kj);
          if (sw) {
            key[i] = kj; key[ixj] = ki;
            int tv = val[i]; val[i] = val[ixj]; val[ixj] = tv;
          }
        }
      }
      __syncthreads();
    }
  }
  for (int i = threadIdx.x; i < NCAND; i += blockDim.x) {
    sScore[(size_t)b * NCAND + i] = key[i];
    sIdx[(size_t)b * NCAND + i]   = val[i];
  }
}

// ---------------- NMS ----------------
__global__ void nms_prep(const float* __restrict__ sScore, const int* __restrict__ sIdx,
                         const float* __restrict__ candBox, const int* __restrict__ candC,
                         float* __restrict__ obox, int* __restrict__ keep) {
  int t = blockIdx.x * blockDim.x + threadIdx.x;
  if (t >= NB * NCAND) return;
  int b = t / NCAND;
  int si = sIdx[t];
  size_t cs = (size_t)b * NCAND + si;
  float off = 4096.f * (float)candC[cs];
  obox[(size_t)t * 4 + 0] = candBox[cs * 4 + 0] + off;
  obox[(size_t)t * 4 + 1] = candBox[cs * 4 + 1] + off;
  obox[(size_t)t * 4 + 2] = candBox[cs * 4 + 2] + off;
  obox[(size_t)t * 4 + 3] = candBox[cs * 4 + 3] + off;
  keep[t] = sScore[t] > 0.05f ? 1 : 0;
}

__global__ __launch_bounds__(1024) void nms_kernel(const float* __restrict__ obox,
                                                   int* __restrict__ keep) {
  int b = blockIdx.x;
  const float* bx = obox + (size_t)b * NCAND * 4;
  int* kp = keep + (size_t)b * NCAND;
  for (int i = 0; i < NCAND - 1; ++i) {
    __syncthreads();                       // also orders prior kp[] writes
    if (kp[i] == 0) continue;              // uniform across block
    float x1 = bx[i * 4], y1 = bx[i * 4 + 1], x2 = bx[i * 4 + 2], y2 = bx[i * 4 + 3];
    float ar = (x2 - x1) * (y2 - y1);
    for (int j = i + 1 + threadIdx.x; j < NCAND; j += blockDim.x) {
      if (!kp[j]) continue;
      float bx1 = bx[j * 4], by1 = bx[j * 4 + 1], bx2 = bx[j * 4 + 2], by2 = bx[j * 4 + 3];
      float xx1 = fmaxf(x1, bx1), yy1 = fmaxf(y1, by1);
      float xx2 = fminf(x2, bx2), yy2 = fminf(y2, by2);
      float w = fmaxf(xx2 - xx1, 0.f), h = fmaxf(yy2 - yy1, 0.f);
      float inter = w * h;
      float arj = (bx2 - bx1) * (by2 - by1);
      float iou = inter / (ar + arj - inter + 1e-9f);
      if (iou > 0.5f) kp[j] = 0;
    }
  }
}

__global__ void write_out(const float* __restrict__ sScore, const int* __restrict__ sIdx,
                          const float* __restrict__ candBox, const int* __restrict__ keep,
                          float* __restrict__ out) {
  int t = blockIdx.x * blockDim.x + threadIdx.x;
  if (t >= NB * NCAND) return;
  int b = t / NCAND;
  int si = sIdx[t];
  float m = keep[t] ? 1.f : 0.f;
  const float* bb = candBox + ((size_t)b * NCAND + si) * 4;
  float* o = out + (size_t)t * 5;
  o[0] = bb[0] * m; o[1] = bb[1] * m; o[2] = bb[2] * m; o[3] = bb[3] * m;
  o[4] = sScore[t] * m;
}

// ---------------- host orchestration ----------------
extern "C" void kernel_launch(void* const* d_in, const int* in_sizes, int n_in,
                              void* d_out, int out_size, void* d_ws, size_t ws_size,
                              hipStream_t stream) {
  (void)in_sizes; (void)n_in; (void)out_size; (void)ws_size;
  static const int Hs[NLEV] = {80, 40, 20, 10, 5};
  static const int Wd[NLEV] = {80, 40, 20, 10, 5};
  static const int Ss[NLEV] = {8, 16, 32, 64, 128};

  const float* feat[NLEV];
  for (int i = 0; i < NLEV; ++i) feat[i] = (const float*)d_in[i];
  const float* cls_w[4]; const float* cls_b[4];
  const float* reg_w[4]; const float* reg_b[4];
  for (int i = 0; i < 4; ++i) {
    cls_w[i] = (const float*)d_in[5 + i];
    cls_b[i] = (const float*)d_in[9 + i];
    reg_w[i] = (const float*)d_in[15 + i];
    reg_b[i] = (const float*)d_in[19 + i];
  }
  const float* cls_out_w = (const float*)d_in[13];
  const float* cls_out_b = (const float*)d_in[14];
  const float* reg_out_w = (const float*)d_in[23];
  const float* reg_out_b = (const float*)d_in[24];

  uint8_t* ws = (uint8_t*)d_ws;
  size_t off = 0;
  auto alloc = [&](size_t bytes) {
    size_t o = off; off += (bytes + 255) & ~(size_t)255; return o;
  };
  const int MAXHW = 80 * 80;
  unsigned short* wtBuf = (unsigned short*)(ws + alloc((size_t)2804 * KTOT * 2));
  unsigned short* actIn = (unsigned short*)(ws + alloc((size_t)NB * MAXPHW * 256 * 2));
  unsigned short* actP  = (unsigned short*)(ws + alloc((size_t)NB * MAXPHW * 256 * 2));
  unsigned short* actQ  = (unsigned short*)(ws + alloc((size_t)NB * MAXPHW * 256 * 2));
  float* probBuf = (float*)(ws + alloc((size_t)NB * MAXHW * 720 * 4));
  float* regBuf  = (float*)(ws + alloc((size_t)NB * MAXHW * 36 * 4));
  float* candS   = (float*)(ws + alloc((size_t)NB * NCAND * 4));
  int*   candI   = (int*)  (ws + alloc((size_t)NB * NCAND * 4));
  float* candBox = (float*)(ws + alloc((size_t)NB * NCAND * 16));
  int*   candC   = (int*)  (ws + alloc((size_t)NB * NCAND * 4));
  float* sScore  = (float*)(ws + alloc((size_t)NB * NCAND * 4));
  int*   sIdx    = (int*)  (ws + alloc((size_t)NB * NCAND * 4));
  float* obox    = (float*)(ws + alloc((size_t)NB * NCAND * 16));
  int*   keep    = (int*)  (ws + alloc((size_t)NB * NCAND * 4));
  int*   tg      = (int*)  (ws + alloc((size_t)NB * 2 * 4));
  int*   cnt     = (int*)  (ws + alloc((size_t)NB * 2 * 4));

  // weight repack (rows: cls 0..1023, cls_out 1024..1743, reg 1744..2767, reg_out 2768..2803)
  auto pack = [&](const float* src, size_t rowOff, int cout) {
    size_t n = (size_t)cout * KTOT;
    pack_weights<<<(unsigned)((n + 255) / 256), 256, 0, stream>>>(
        src, wtBuf + rowOff * KTOT, cout);
  };
  for (int i = 0; i < 4; ++i) pack(cls_w[i], (size_t)i * 256, 256);
  pack(cls_out_w, 1024, 720);
  for (int i = 0; i < 4; ++i) pack(reg_w[i], 1744 + (size_t)i * 256, 256);
  pack(reg_out_w, 2768, 36);

  zero_u32<<<64, 256, 0, stream>>>((unsigned*)candS, (size_t)NB * NCAND);
  zero_u32<<<64, 256, 0, stream>>>((unsigned*)candI, (size_t)NB * NCAND);
  zero_u32<<<64, 256, 0, stream>>>((unsigned*)candBox, (size_t)NB * NCAND * 4);
  zero_u32<<<64, 256, 0, stream>>>((unsigned*)candC, (size_t)NB * NCAND);

  for (int lv = 0; lv < NLEV; ++lv) {
    int H = Hs[lv], W = Wd[lv], HW = H * W;
    int PH = H + 2, PW = W + 2;
    // zero halos of the two tower ping-pong buffers (interiors get overwritten)
    size_t padded_u32 = (size_t)NB * PH * PW * 128;   // ushort count / 2
    zero_u32<<<512, 256, 0, stream>>>((unsigned*)actP, padded_u32);
    zero_u32<<<512, 256, 0, stream>>>((unsigned*)actQ, padded_u32);

    size_t tot = (size_t)NB * PH * PW * 256;
    nchw_to_padded<<<(unsigned)((tot + 255) / 256), 256, 0, stream>>>(
        feat[lv], actIn, H, W);

    auto conv = [&](const unsigned short* in, size_t wRow, const float* bias,
                    unsigned short* ob, float* of, int Cout, int mode) {
      int nT = (Cout + 15) / 16;
      dim3 g((unsigned)((HW + 15) / 16), (unsigned)((nT + 7) / 8), NB);
      conv3x3_wmma<<<g, 256, 0, stream>>>(in, wtBuf + wRow * (size_t)KTOT, bias,
                                          ob, of, H, W, Cout, nT, mode);
    };
    // classification tower
    conv(actIn, 0,    cls_b[0], actP, nullptr, 256, 0);
    conv(actP,  256,  cls_b[1], actQ, nullptr, 256, 0);
    conv(actQ,  512,  cls_b[2], actP, nullptr, 256, 0);
    conv(actP,  768,  cls_b[3], actQ, nullptr, 256, 0);
    conv(actQ,  1024, cls_out_b, nullptr, probBuf, 720, 1);   // fused sigmoid
    // regression tower
    conv(actIn, 1744, reg_b[0], actP, nullptr, 256, 0);
    conv(actP,  2000, reg_b[1], actQ, nullptr, 256, 0);
    conv(actQ,  2256, reg_b[2], actP, nullptr, 256, 0);
    conv(actP,  2512, reg_b[3], actQ, nullptr, 256, 0);
    conv(actQ,  2768, reg_out_b, nullptr, regBuf, 36, 2);

    int N = HW * 720;
    zero_u32<<<1, 64, 0, stream>>>((unsigned*)cnt, (size_t)NB * 2);
    hist_topk<<<NB, 1024, 0, stream>>>(probBuf, N, tg);
    unsigned gx = (unsigned)((N + 255) / 256); if (gx > 2048u) gx = 2048u;
    dim3 cg(gx, NB);
    topk_compact<<<cg, 256, 0, stream>>>(probBuf, tg, cnt, candS, candI, N, lv);
    decode_boxes<<<(NB * TOPK_N + 255) / 256, 256, 0, stream>>>(
        candI, regBuf, candBox, candC, H, W, Ss[lv], lv);
  }

  sort_desc<<<NB, 1024, SORT_N * 8, stream>>>(candS, sScore, sIdx);
  nms_prep<<<(NB * NCAND + 255) / 256, 256, 0, stream>>>(
      sScore, sIdx, candBox, candC, obox, keep);
  nms_kernel<<<NB, 1024, 0, stream>>>(obox, keep);
  write_out<<<(NB * NCAND + 255) / 256, 256, 0, stream>>>(
      sScore, sIdx, candBox, keep, (float*)d_out);
}
